// TensorProductScoreModel_19894288515356
// MI455X (gfx1250) — matrix operand order repaired and checked by reading, not compiled, not run on hardware
//
#include <hip/hip_runtime.h>
#include <hip/hip_bf16.h>
#include <math.h>

#define NNODES 10000
#define NEDGES 40000
#define DMAX   112
#define EPSBN  1e-5f
#define SIG_MIN (0.01f*3.14159265358979f)
#define SIG_RATIO_LOG 4.6051701859880914f   /* log(100) */

typedef __attribute__((ext_vector_type(16))) _Float16 v16h;
typedef __attribute__((ext_vector_type(8)))  float    v8f;

// ---------------- Tensor-product path tables (compile-time constants) -------
// cg: 0=(0,0,0) 1=(0,1,1) 2=(1,0,1) 3=(1,1,0) 4=(1,1,1)
struct TPPath { int cg, in_off, m1, out_off, m3, w_off; float alpha; };

__constant__ TPPath g_paths[25] = {
  // Layer 0: in [(32,0)], out [(32,0),(8,1)]
  {0, 0,32,  0,32,    0, 0.17677670f},
  {1, 0,32, 32, 8, 1024, 0.30618622f},
  // Layer 1: in [(32,0),(8,1)], out [(32,0),(8,1),(8,1)]
  {0, 0,32,  0,32,    0, 0.15811388f},
  {1, 0,32, 32, 8, 1024, 0.27386128f},
  {2,32, 8, 32, 8, 1280, 0.27386128f},
  {3,32, 8,  0,32, 1344, 0.15811388f},
  {4,32, 8, 56, 8, 1600, 0.61237244f},
  // Layer 2: in SEQ2, out SEQ3
  {0, 0,32,  0,32,    0, 0.15811388f},
  {1, 0,32, 32, 8, 1024, 0.25f},
  {2,32, 8, 32, 8, 1280, 0.25f},
  {3,32, 8,  0,32, 1344, 0.15811388f},
  {4,32, 8, 56, 8, 1600, 0.43301270f},
  {2,56, 8, 56, 8, 1664, 0.43301270f},
  {4,56, 8, 32, 8, 1728, 0.25f},
  {3,56, 8, 80,32, 1792, 0.35355339f},
  // Layer 3: in SEQ3, out SEQ3
  {0, 0,32,  0,32,    0, 0.15811388f},
  {1, 0,32, 32, 8, 1024, 0.25f},
  {2,32, 8, 32, 8, 1280, 0.25f},
  {3,32, 8,  0,32, 1344, 0.15811388f},
  {4,32, 8, 56, 8, 1600, 0.25f},
  {2,56, 8, 56, 8, 1664, 0.25f},
  {4,56, 8, 32, 8, 1728, 0.25f},
  {3,56, 8, 80,32, 1792, 0.15811388f},
  {0,80,32, 80,32, 2048, 0.15811388f},
  {1,80,32, 56, 8, 3072, 0.25f},
};
__constant__ int g_poff[4] = {0, 2, 7, 15};
__constant__ int g_pcnt[4] = {2, 5, 8, 10};

// ---------------- small utility kernels -------------------------------------
__global__ void k_zero(float* p, int n) {
  int i = blockIdx.x*blockDim.x + threadIdx.x;
  if (i < n) p[i] = 0.0f;
}
__global__ void k_addres(float* __restrict__ acc, const float* __restrict__ na, int n) {
  int i = blockIdx.x*blockDim.x + threadIdx.x;
  if (i < n) acc[i] += na[i];
}
__global__ void k_copy(const float* __restrict__ src, float* __restrict__ dst, int n) {
  int i = blockIdx.x*blockDim.x + threadIdx.x;
  if (i < n) dst[i] = src[i];
}

// ---- pack f32 row-major K x N weights into WMMA B-fragment order (f16) -----
// Bpk[((tn*KT + kt)*32 + lane)*16 + h]  <=  B[(kt*32 + 16*(h>=8) + 8*(lane>=16) + (h&7)) * N + tn*16 + (lane&15)]
__global__ void k_packB(const float* __restrict__ in, _Float16* __restrict__ out,
                        int K, int N) {
  int o = blockIdx.x*blockDim.x + threadIdx.x;
  if (o >= K*N) return;
  int KT = K >> 5;
  int h    = o & 15;
  int lane = (o >> 4) & 31;
  int rest = o >> 9;
  int kt = rest % KT;
  int tn = rest / KT;
  int k = kt*32 + ((h & 8) ? 16 : 0) + ((lane & 16) ? 8 : 0) + (h & 7);
  int n = tn*16 + (lane & 15);
  out[o] = (_Float16)in[(size_t)k*N + n];
}

// ---------------- node preprocessing (wave per node, lane = channel) --------
__global__ void k_node_pre(const float* __restrict__ x, const float* __restrict__ sigma,
                           const float* __restrict__ w1, const float* __restrict__ b1,
                           const float* __restrict__ w2, const float* __restrict__ b2,
                           float* __restrict__ semb, float* __restrict__ na) {
  int wave = (blockIdx.x*blockDim.x + threadIdx.x) >> 5;
  int lane = threadIdx.x & 31;
  if (wave >= NNODES) return;
  int nd = wave;
  float t = logf(sigma[nd] / SIG_MIN) * (10000.0f / SIG_RATIO_LOG);
  float freq = expf((float)(lane & 15) * (-logf(10000.0f) / 15.0f));
  float my_semb = (lane < 16) ? sinf(t*freq) : cosf(t*freq);
  semb[nd*32 + lane] = my_semb;

  float h = b1[lane];
  for (int k = 0; k < 74; ++k)
    h += x[nd*74 + k] * w1[k*32 + lane];
  for (int k = 0; k < 32; ++k)
    h += __shfl(my_semb, k, 32) * w1[(74 + k)*32 + lane];
  h = h > 0.0f ? h : 0.0f;

  float o = b2[lane];
  for (int k = 0; k < 32; ++k)
    o += __shfl(h, k, 32) * w2[k*32 + lane];

  size_t base = (size_t)nd * DMAX;
  na[base + lane]      = o;
  na[base + 32 + lane] = 0.0f;
  na[base + 64 + lane] = 0.0f;
  if (lane < 16) na[base + 96 + lane] = 0.0f;
}

// ---------------- edge preprocessing (wave per edge, lane = channel) --------
__global__ void k_edge_pre(const float* __restrict__ pos, const float* __restrict__ eattr,
                           const float* __restrict__ semb, const int* __restrict__ ei,
                           const float* __restrict__ w1, const float* __restrict__ b1,
                           const float* __restrict__ w2, const float* __restrict__ b2,
                           float* __restrict__ vhat, float* __restrict__ eemb,
                           float* __restrict__ cnt) {
  int wave = (blockIdx.x*blockDim.x + threadIdx.x) >> 5;
  int lane = threadIdx.x & 31;
  if (wave >= NEDGES) return;
  int e = wave;
  int src = ei[e], dst = ei[NEDGES + e];
  float vx = pos[dst*3+0] - pos[src*3+0];
  float vy = pos[dst*3+1] - pos[src*3+1];
  float vz = pos[dst*3+2] - pos[src*3+2];
  float d = sqrtf(vx*vx + vy*vy + vz*vz);
  d = fmaxf(d, 1e-6f);
  float v0 = vx/d, v1 = vy/d, v2 = vz/d;
  if (lane == 0) {
    vhat[e*3+0] = v0; vhat[e*3+1] = v1; vhat[e*3+2] = v2;
    atomicAdd(&cnt[dst], 1.0f);
  }
  // eattr element k of 86: [edge_attr(4), semb[src](32), rbf(50)]
  const float step = 5.0f / 49.0f;
  const float coeff = -0.5f / (step*step);
  auto eav = [&](int k)->float {
    if (k < 4)  return eattr[e*4 + k];
    if (k < 36) return semb[src*32 + (k - 4)];
    float offv = (float)(k - 36) * step;
    float dd = d - offv;
    return expf(coeff * dd * dd);
  };
  float ea0 = eav(lane);
  float ea1 = eav(lane + 32);
  float ea2 = (lane + 64 < 86) ? eav(lane + 64) : 0.0f;

  float h = b1[lane];
  for (int k = 0; k < 86; ++k) {
    float src_v = (k < 32) ? ea0 : (k < 64) ? ea1 : ea2;
    float val = __shfl(src_v, k & 31, 32);
    h += val * w1[k*32 + lane];
  }
  h = h > 0.0f ? h : 0.0f;
  float o = b2[lane];
  for (int k = 0; k < 32; ++k)
    o += __shfl(h, k, 32) * w2[k*32 + lane];
  eemb[e*32 + lane] = o;
}

// ---------------- build f16 eac = [eemb | na[src,:32] | na[dst,:32]] --------
__global__ void k_build_eac(const float* __restrict__ eemb, const float* __restrict__ na,
                            const int* __restrict__ ei, _Float16* __restrict__ eac) {
  int i = blockIdx.x*blockDim.x + threadIdx.x;
  if (i >= NEDGES*96) return;
  int e = i / 96, c = i - e*96;
  float v;
  if (c < 32)       v = eemb[e*32 + c];
  else if (c < 64)  v = na[(size_t)ei[e]*DMAX + (c - 32)];
  else              v = na[(size_t)ei[NEDGES + e]*DMAX + (c - 64)];
  eac[i] = (_Float16)v;
}

// ---------------- WMMA GEMM core --------------------------------------------
// A: row-major M x K (f16), K = 32*KT compile-time.  Bpk: fragment-packed.
// Each wave computes a 16 x (16*NT) strip.  All fragments are preloaded with
// immediate-offset b128 clauses from two base pointers, then WMMAs issue
// back-to-back so loads overlap the matrix pipe.
template <int NT, int KT>
__device__ __forceinline__ void wmma_strip(const _Float16* __restrict__ A,
                                           const _Float16* __restrict__ Bpk,
                                           int tm, int tn0, int lane, v8f* c) {
  const _Float16* Ab = A + (size_t)(tm*16 + (lane & 15))*(KT*32) + ((lane & 16) ? 8 : 0);
  const _Float16* Bb = Bpk + ((size_t)tn0*KT*32 + lane)*16;
  v16h a[KT];
  v16h b[NT*KT];
  #pragma unroll
  for (int kt = 0; kt < KT; ++kt) {
    #pragma unroll
    for (int h = 0; h < 8; ++h) {
      a[kt][h]     = Ab[kt*32 + h];
      a[kt][8 + h] = Ab[kt*32 + 16 + h];
    }
  }
  #pragma unroll
  for (int j = 0; j < NT; ++j)
    #pragma unroll
    for (int kt = 0; kt < KT; ++kt)
      b[j*KT + kt] = *(const v16h*)(Bb + (size_t)(j*KT + kt)*512);
  #pragma unroll
  for (int kt = 0; kt < KT; ++kt)
    #pragma unroll
    for (int j = 0; j < NT; ++j)
      c[j] = __builtin_amdgcn_wmma_f32_16x16x32_f16(false, a[kt], false, b[j*KT + kt],
                                                    (short)0, c[j], false, false);
}

// NT=3, relu, f16 out  (N == 96, K == 96)
__global__ void k_gemm_h(const _Float16* __restrict__ A, const _Float16* __restrict__ Bpk,
                         const float* __restrict__ bias, _Float16* __restrict__ C,
                         int M, int N) {
  int gid = blockIdx.x*blockDim.x + threadIdx.x;
  int wave = gid >> 5, lane = gid & 31;
  const int NT = 3;
  int groups = (N >> 4) / NT;
  int tm = wave / groups, tn0 = (wave - tm*groups) * NT;
  if (tm*16 >= M) return;           // wave-uniform: EXEC stays all-ones for WMMA
  v8f c[NT] = {};
  wmma_strip<NT, 3>(A, Bpk, tm, tn0, lane, c);
  int mbase = tm*16 + ((lane & 16) ? 8 : 0);
  #pragma unroll
  for (int j = 0; j < NT; ++j) {
    int nc = (tn0 + j)*16 + (lane & 15);
    float bv = bias[nc];
    #pragma unroll
    for (int r = 0; r < 8; ++r) {
      float v = c[j][r] + bv;
      v = v > 0.0f ? v : 0.0f;
      C[(size_t)(mbase + r)*N + nc] = (_Float16)v;
    }
  }
}

// NT=4, no relu, f32 out  (N multiple of 64, K == 96)
__global__ void k_gemm_w(const _Float16* __restrict__ A, const _Float16* __restrict__ Bpk,
                         const float* __restrict__ bias, float* __restrict__ C,
                         int M, int N) {
  int gid = blockIdx.x*blockDim.x + threadIdx.x;
  int wave = gid >> 5, lane = gid & 31;
  const int NT = 4;
  int groups = (N >> 4) / NT;
  int tm = wave / groups, tn0 = (wave - tm*groups) * NT;
  if (tm*16 >= M) return;
  v8f c[NT] = {};
  wmma_strip<NT, 3>(A, Bpk, tm, tn0, lane, c);
  int mbase = tm*16 + ((lane & 16) ? 8 : 0);
  #pragma unroll
  for (int j = 0; j < NT; ++j) {
    int nc = (tn0 + j)*16 + (lane & 15);
    float bv = bias[nc];
    #pragma unroll
    for (int r = 0; r < 8; ++r)
      C[(size_t)(mbase + r)*N + nc] = c[j][r] + bv;
  }
}

// ---------------- tensor product + scatter (wave per edge) ------------------
__global__ void k_tp(const float* __restrict__ wbuf, const float* __restrict__ na,
                     const float* __restrict__ vhat, const int* __restrict__ ei,
                     const float* __restrict__ cnt, float* __restrict__ tpacc,
                     int layer, int nW, int e0, int nE) {
  int wave = (blockIdx.x*blockDim.x + threadIdx.x) >> 5;
  int lane = threadIdx.x & 31;
  if (wave >= nE) return;
  int e = e0 + wave;
  int src = ei[e], dst = ei[NEDGES + e];
  const float* xsrc = na + (size_t)src * DMAX;
  float vh[3] = { vhat[e*3+0], vhat[e*3+1], vhat[e*3+2] };
  float cinv = 1.0f / fmaxf(cnt[dst], 1.0f);
  const float* w = wbuf + (size_t)wave * nW;

  int p0 = g_poff[layer], pc = g_pcnt[layer];
  for (int p = p0; p < p0 + pc; ++p) {
    TPPath pa = g_paths[p];
    int deg = (pa.cg == 0 || pa.cg == 3) ? 1 : 3;   // 2*l3+1
    int nOut = pa.m3 * deg;
    if (lane < nOut) {
      int u3 = lane / deg, k = lane - u3*deg;
      float acc = 0.0f;
      for (int u1 = 0; u1 < pa.m1; ++u1) {
        float t;
        if (pa.cg == 0) {
          t = xsrc[pa.in_off + u1];
        } else if (pa.cg == 1) {
          t = xsrc[pa.in_off + u1] * vh[k];
        } else if (pa.cg == 2) {
          t = xsrc[pa.in_off + u1*3 + k] * 0.57735027f;
        } else if (pa.cg == 3) {
          const float* xv = xsrc + pa.in_off + u1*3;
          t = xv[0]*vh[0] + xv[1]*vh[1] + xv[2]*vh[2];
        } else { // cg == 4: cross product / sqrt(2)
          const float* xv = xsrc + pa.in_off + u1*3;
          int k1 = (k + 1) % 3, k2 = (k + 2) % 3;
          t = (xv[k1]*vh[k2] - xv[k2]*vh[k1]) * 0.70710678f;
        }
        acc += t * w[pa.w_off + u1*pa.m3 + u3];
      }
      atomicAdd(&tpacc[(size_t)dst*DMAX + pa.out_off + lane], pa.alpha * acc * cinv);
    }
  }
}

// ---------------- equivariant batch norm ------------------------------------
__global__ void k_bn_stats(const float* __restrict__ acc, float* __restrict__ S1,
                           float* __restrict__ S2, int dout) {
  int c = blockIdx.x;
  if (c >= dout) return;
  __shared__ float s1[256], s2[256];
  float a1 = 0.0f, a2 = 0.0f;
  for (int n = threadIdx.x; n < NNODES; n += blockDim.x) {
    float v = acc[(size_t)n*DMAX + c];
    a1 += v; a2 += v*v;
  }
  s1[threadIdx.x] = a1; s2[threadIdx.x] = a2;
  __syncthreads();
  for (int s = 128; s > 0; s >>= 1) {
    if ((int)threadIdx.x < s) { s1[threadIdx.x] += s1[threadIdx.x+s]; s2[threadIdx.x] += s2[threadIdx.x+s]; }
    __syncthreads();
  }
  if (threadIdx.x == 0) { S1[c] = s1[0]; S2[c] = s2[0]; }
}

__global__ void k_bn_apply(const float* __restrict__ acc, const float* __restrict__ S1,
                           const float* __restrict__ S2, float* __restrict__ na, int dout) {
  int i = blockIdx.x*blockDim.x + threadIdx.x;
  if (i >= NNODES*DMAX) return;
  int c = i % DMAX;
  float out = 0.0f;
  if (c < dout) {
    float v = acc[i];
    const float invN = 1.0f / (float)NNODES;
    if (c < 32 || c >= 80) {           // scalar (l=0) channels
      float mean = S1[c] * invN;
      float var  = S2[c] * invN - mean*mean;
      out = (v - mean) * rsqrtf(var + EPSBN);
    } else {                           // vector (l=1) channels, groups of 3 from 32
      int g = 32 + ((c - 32) / 3) * 3;
      float n2 = (S2[g] + S2[g+1] + S2[g+2]) * invN * (1.0f/3.0f);
      out = v * rsqrtf(n2 + EPSBN);
    }
  }
  na[i] = out;
}

// ---------------- host orchestration ----------------------------------------
extern "C" void kernel_launch(void* const* d_in, const int* in_sizes, int n_in,
                              void* d_out, int out_size, void* d_ws, size_t ws_size,
                              hipStream_t stream) {
  (void)in_sizes; (void)n_in;
  const float* x        = (const float*)d_in[0];
  const float* pos      = (const float*)d_in[1];
  const float* sigma    = (const float*)d_in[2];
  const float* eattr    = (const float*)d_in[3];
  const float* node_w1  = (const float*)d_in[4];
  const float* node_b1  = (const float*)d_in[5];
  const float* node_w2  = (const float*)d_in[6];
  const float* node_b2  = (const float*)d_in[7];
  const float* edge_w1  = (const float*)d_in[8];
  const float* edge_b1  = (const float*)d_in[9];
  const float* edge_w2  = (const float*)d_in[10];
  const float* edge_b2  = (const float*)d_in[11];
  const float* fc1_w    = (const float*)d_in[12];
  const float* fc1_b    = (const float*)d_in[13];
  const float* fc2_w[4] = {(const float*)d_in[14], (const float*)d_in[16],
                           (const float*)d_in[18], (const float*)d_in[20]};
  const float* fc2_b[4] = {(const float*)d_in[15], (const float*)d_in[17],
                           (const float*)d_in[19], (const float*)d_in[21]};
  const int*   ei       = (const int*)d_in[22];

  static const int nW[4]   = {1280, 1664, 2048, 3328};
  static const int dout[4] = {56, 80, 112, 112};

  char* base = (char*)d_ws;
  size_t off = 0;
  auto alloc = [&](size_t bytes) -> char* {
    char* p = base + off;
    off = (off + bytes + 255) & ~(size_t)255;
    return p;
  };
  float*     semb  = (float*)alloc((size_t)NNODES*32*4);
  float*     na    = (float*)alloc((size_t)NNODES*DMAX*4);
  float*     tpacc = (float*)alloc((size_t)NNODES*DMAX*4);
  float*     vhat  = (float*)alloc((size_t)NEDGES*3*4);
  float*     eemb  = (float*)alloc((size_t)NEDGES*32*4);
  float*     cnt   = (float*)alloc((size_t)NNODES*4);
  float*     S1    = (float*)alloc(DMAX*4);
  float*     S2    = (float*)alloc(DMAX*4);
  _Float16*  eac16 = (_Float16*)alloc((size_t)NEDGES*96*2);
  _Float16*  h16   = (_Float16*)alloc((size_t)NEDGES*96*2);
  _Float16*  W1p   = (_Float16*)alloc((size_t)96*96*2);
  _Float16*  W2p   = (_Float16*)alloc((size_t)96*3328*2);
  float*     wbuf  = (float*)(base + off);
  size_t avail = (ws_size > off) ? (ws_size - off) : 0;

  k_zero<<<(NNODES + 255)/256, 256, 0, stream>>>(cnt, NNODES);
  k_node_pre<<<(NNODES + 3)/4, 128, 0, stream>>>(x, sigma, node_w1, node_b1,
                                                 node_w2, node_b2, semb, na);
  k_edge_pre<<<(NEDGES + 3)/4, 128, 0, stream>>>(pos, eattr, semb, ei,
                                                 edge_w1, edge_b1, edge_w2, edge_b2,
                                                 vhat, eemb, cnt);

  for (int L = 0; L < 4; ++L) {
    int n = nW[L];
    k_packB<<<(96*96 + 255)/256, 256, 0, stream>>>(fc1_w + (size_t)L*96*96, W1p, 96, 96);
    k_packB<<<(96*n + 255)/256, 256, 0, stream>>>(fc2_w[L], W2p, 96, n);
    k_build_eac<<<(NEDGES*96 + 255)/256, 256, 0, stream>>>(eemb, na, ei, eac16);

    {
      int waves = (NEDGES/16) * ((96/16)/3);
      k_gemm_h<<<(waves*32 + 255)/256, 256, 0, stream>>>(eac16, W1p, fc1_b + L*96,
                                                         h16, NEDGES, 96);
    }

    k_zero<<<(NNODES*DMAX + 255)/256, 256, 0, stream>>>(tpacc, NNODES*DMAX);

    long chunk = (long)(avail / ((size_t)n * 4));
    chunk &= ~15L;
    if (chunk < 16) chunk = 16;
    if (chunk > NEDGES) chunk = NEDGES;
    for (int s0 = 0; s0 < NEDGES; s0 += (int)chunk) {
      int rows = NEDGES - s0;
      if (rows > (int)chunk) rows = (int)chunk;
      int waves = (rows/16) * ((n/16)/4);
      k_gemm_w<<<(waves*32 + 255)/256, 256, 0, stream>>>(h16 + (size_t)s0*96, W2p,
                                                         fc2_b[L], wbuf, rows, n);
      k_tp<<<(rows + 3)/4, 128, 0, stream>>>(wbuf, na, vhat, ei, cnt, tpacc,
                                             L, n, s0, rows);
    }

    k_addres<<<(NNODES*DMAX + 255)/256, 256, 0, stream>>>(tpacc, na, NNODES*DMAX);
    k_bn_stats<<<dout[L], 256, 0, stream>>>(tpacc, S1, S2, dout[L]);
    k_bn_apply<<<(NNODES*DMAX + 255)/256, 256, 0, stream>>>(tpacc, S1, S2, na, dout[L]);
  }

  int ncopy = out_size < NNODES*DMAX ? out_size : NNODES*DMAX;
  k_copy<<<(ncopy + 255)/256, 256, 0, stream>>>(na, (float*)d_out, ncopy);
}
